// VAE_75127567941633
// MI455X (gfx1250) — compile-verified, matrix operand order checked by reading, not run on hardware
//
#include <hip/hip_runtime.h>
#include <hip/hip_bf16.h>
#include <math.h>

typedef __bf16 bf16;
typedef __attribute__((ext_vector_type(16))) __bf16 v16bf;
typedef __attribute__((ext_vector_type(8)))  __bf16 v8bf;
typedef __attribute__((ext_vector_type(8)))  float  v8f;
typedef __attribute__((ext_vector_type(2)))  float  v2f;

#define TFRM 8192
#define LINGD 442
#define ACD   199
#define CATD  641
#define CATP  672
#define HIDD  400
#define G4H   1600
#define ZD    16
#define X2D   458
#define X2P   480
#define H2D   800
#define NMOR  1024

static __host__ __device__ inline int npad64(int n) { return (n + 63) & ~63; }

// packed fp32 FMA (VOP3P V_PK_FMA_F32): 2 fp32 FMAs per instruction
__device__ inline v2f pk_fma_f32(v2f a, v2f b, v2f c)
{
    v2f d;
    asm("v_pk_fma_f32 %0, %1, %2, %3" : "=v"(d) : "v"(a), "v"(b), "v"(c));
    return d;
}

// ------------------------------------------------------------------
// WMMA GEMM:  C[M,N] = act(A[M,K] @ B[Npad,K]^T + bias)
// A: bf16 [M, lda], K padded with zeros.  B: bf16 [Npad64, ldb], rows >= N
// zero-filled  =>  the k-loop has NO guards: pure load + wmma.
// One wave computes a 32x64 C block (2 M-tiles x 4 N-subtiles):
// 8 WMMAs per 12 b128-loads per k-step.
// OUT_BF16: writes the full padded width ldc (zeros for n>=N) so the
// output can be re-used directly as a padded A matrix downstream.
// ------------------------------------------------------------------
template<bool OUT_BF16, bool RELU>
__global__ __launch_bounds__(32)
void gemm_bias_wmma(const bf16* __restrict__ A, const bf16* __restrict__ B,
                    const float* __restrict__ bias, void* __restrict__ Cv,
                    int N, int K, int lda, int ldb, int ldc)
{
    const int lane = threadIdx.x;
    const int m0 = blockIdx.y << 5;         // 2 M-tiles of 16
    const int n0 = blockIdx.x << 6;         // 4 N-subtiles of 16
    const int row  = lane & 15;
    const int hi   = lane >> 4;
    const int off0 = hi ? 8 : 0;            // 16-bit A-matrix 16x32 lane layout (ISA 7.12.2)
    const int off1 = hi ? 24 : 16;

    const bf16* arow0 = A + (size_t)(m0 + row) * lda;
    const bf16* arow1 = arow0 + (size_t)16 * lda;
    const bf16* brow  = B + (size_t)(n0 + row) * ldb;
    const size_t bstep = (size_t)16 * ldb;

    v8f acc[2][4] = {};
    union U { v16bf v; v8bf h[2]; };
    for (int k0 = 0; k0 < K; k0 += 32) {
        U ua0, ua1;
        ua0.h[0] = *(const v8bf*)(arow0 + k0 + off0);
        ua0.h[1] = *(const v8bf*)(arow0 + k0 + off1);
        ua1.h[0] = *(const v8bf*)(arow1 + k0 + off0);
        ua1.h[1] = *(const v8bf*)(arow1 + k0 + off1);
        const bf16* bp = brow + k0;
        U ub[4];
#pragma unroll
        for (int s = 0; s < 4; ++s) {
            ub[s].h[0] = *(const v8bf*)(bp + s * bstep + off0);
            ub[s].h[1] = *(const v8bf*)(bp + s * bstep + off1);
        }
#pragma unroll
        for (int s = 0; s < 4; ++s) {
            acc[0][s] = __builtin_amdgcn_wmma_f32_16x16x32_bf16(
                            false, ua0.v, false, ub[s].v, (short)0, acc[0][s], false, false);
            acc[1][s] = __builtin_amdgcn_wmma_f32_16x16x32_bf16(
                            false, ua1.v, false, ub[s].v, (short)0, acc[1][s], false, false);
        }
    }
#pragma unroll
    for (int s = 0; s < 4; ++s) {
        const int n = n0 + (s << 4) + row;  // C layout: col = lane&15
        const bool nok = n < N;
        const float badd = nok ? bias[n] : 0.0f;
#pragma unroll
        for (int mt = 0; mt < 2; ++mt) {
#pragma unroll
            for (int r = 0; r < 8; ++r) {   // C layout: row = r + 8*(lane>=16)
                int m = m0 + (mt << 4) + r + (hi << 3);
                float v = nok ? acc[mt][s][r] + badd : 0.0f;
                if (RELU) v = fmaxf(v, 0.0f);
                if (OUT_BF16) {
                    if (n < ldc) ((bf16*)Cv)[(size_t)m * ldc + n] = (bf16)v;
                } else if (nok) {
                    ((float*)Cv)[(size_t)m * ldc + n] = v;
                }
            }
        }
    }
}

// ---- weight fp32 [N,K] -> bf16 [Npad,Kpad], zero pads both dims ----
__global__ void k_cvt_pad(const float* __restrict__ src, bf16* __restrict__ dst,
                          int N, int K, int Npad, int Kpad)
{
    int i = blockIdx.x * blockDim.x + threadIdx.x;
    if (i >= Npad * Kpad) return;
    int r = i / Kpad, c = i - r * Kpad;
    dst[i] = (r < N && c < K) ? (bf16)src[r * K + c] : (bf16)0.0f;
}

__global__ void k_addbias(const float* a, const float* b, float* o, int n)
{
    int i = blockIdx.x * blockDim.x + threadIdx.x;
    if (i < n) o[i] = a[i] + b[i];
}

// concat(ling, ac) -> bf16 [T, CATP] (pads zero)
__global__ void k_concat_in(const float* __restrict__ ling, const float* __restrict__ ac,
                            bf16* __restrict__ dst)
{
    int i = blockIdx.x * blockDim.x + threadIdx.x;
    if (i >= TFRM * CATP) return;
    int t = i / CATP, c = i - t * CATP;
    float v = 0.0f;
    if (c < LINGD)      v = ling[t * LINGD + c];
    else if (c < CATD)  v = ac[t * ACD + (c - LINGD)];
    dst[i] = (bf16)v;
}

// concat(ling, z_broadcast) -> bf16 [T, X2P]
__global__ void k_build_x2(const float* __restrict__ ling, const float* __restrict__ z,
                           const int* __restrict__ kcum, bf16* __restrict__ dst)
{
    int i = blockIdx.x * blockDim.x + threadIdx.x;
    if (i >= TFRM * X2P) return;
    int t = i / X2P, c = i - t * X2P;
    float v = 0.0f;
    if (c < LINGD) {
        v = ling[t * LINGD + c];
    } else if (c < X2D) {
        int k = kcum[t];                 // inclusive cumsum of mora_index
        if (k < NMOR) v = z[k * ZD + (c - LINGD)];
    }
    dst[i] = (bf16)v;
}

// concat(hs_f, hs_b) -> bf16 [T, H2D], optional relu
template<bool RELU>
__global__ void k_concat_hs(const float* __restrict__ f, const float* __restrict__ b,
                            bf16* __restrict__ dst)
{
    int i = blockIdx.x * blockDim.x + threadIdx.x;
    if (i >= TFRM * H2D) return;
    int t = i / H2D, c = i - t * H2D;
    float v = (c < HIDD) ? f[(size_t)t * HIDD + c] : b[(size_t)t * HIDD + (c - HIDD)];
    if (RELU) v = fmaxf(v, 0.0f);
    dst[i] = (bf16)v;
}

// inclusive prefix scan of mora_index + boundary index list
__global__ void k_mora_scan(const float* __restrict__ mora, int* __restrict__ kcum,
                            int* __restrict__ idx)
{
    __shared__ int sums[1024];
    int tid = threadIdx.x;
    const int per = TFRM / 1024;        // 8
    int base = tid * per;
    int loc[per];
    int s = 0;
#pragma unroll
    for (int i = 0; i < per; ++i) { loc[i] = mora[base + i] > 0.5f ? 1 : 0; s += loc[i]; }
    sums[tid] = s;
    __syncthreads();
    if (tid == 0) {
        int acc = 0;
        for (int i = 0; i < 1024; ++i) { int v = sums[i]; sums[i] = acc; acc += v; }
    }
    __syncthreads();
    int acc = sums[tid];
#pragma unroll
    for (int i = 0; i < per; ++i) {
        acc += loc[i];
        kcum[base + i] = acc;
        if (loc[i]) { int j = acc - 1; if (j < NMOR) idx[j] = base + i; }
    }
}

// gather relu(out1[idx]) -> mu/logvar -> z ; write mu/logvar to output
__global__ void k_gather_fc2(const float* __restrict__ hsf, const float* __restrict__ hsb,
                             const int* __restrict__ idx,
                             const float* __restrict__ W21, const float* __restrict__ b21,
                             const float* __restrict__ W22, const float* __restrict__ b22,
                             const float* __restrict__ eps,
                             float* __restrict__ z, float* __restrict__ out_mu,
                             float* __restrict__ out_lv)
{
    __shared__ __align__(16) float h[H2D];
    __shared__ float g32[32];
    int m = blockIdx.x;
    int t = idx[m];
    for (int j = threadIdx.x; j < H2D; j += blockDim.x) {
        float v = (j < HIDD) ? hsf[(size_t)t * HIDD + j] : hsb[(size_t)t * HIDD + (j - HIDD)];
        h[j] = fmaxf(v, 0.0f);
    }
    __syncthreads();
    if (threadIdx.x < 32) {
        int o = threadIdx.x;            // 0..15 mu, 16..31 logvar
        int c = o & 15;
        const float* W = (o < 16) ? (W21 + c * H2D) : (W22 + c * H2D);
        float acc = (o < 16) ? b21[c] : b22[c];
        for (int j = 0; j < H2D; ++j) acc = fmaf(W[j], h[j], acc);
        g32[o] = acc;
        if (o < 16) out_mu[m * ZD + c] = acc;
        else        out_lv[m * ZD + c] = acc;
    }
    __syncthreads();
    if (threadIdx.x < 16) {
        int c = threadIdx.x;
        float mu = g32[c], lv = g32[16 + c];
        z[m * ZD + c] = mu + eps[m * ZD + c] * expf(0.5f * lv);
    }
}

// ------------------- LSTM recurrence (one block per direction) -----
// Per step: g[1600] = xs[t] + Whh @ h  via packed fp32 FMAs
// (v_pk_fma_f32: 2 FMA/instr, two independent chains), then gate math.
__global__ __launch_bounds__(1024)
void k_lstm_scan(const float* __restrict__ xs_f, const float* __restrict__ xs_b,
                 const float* __restrict__ Whh_f, const float* __restrict__ Whh_b,
                 float* __restrict__ hs_f, float* __restrict__ hs_b)
{
    const int rev = blockIdx.x;
    const float* xs  = rev ? xs_b  : xs_f;
    const float* Whh = rev ? Whh_b : Whh_f;
    float* hs        = rev ? hs_b  : hs_f;

    __shared__ __align__(16) float h[HIDD];
    __shared__ float c[HIDD];
    __shared__ float g[G4H];
    const int tid = threadIdx.x;
    for (int j = tid; j < HIDD; j += 1024) { h[j] = 0.0f; c[j] = 0.0f; }
    __syncthreads();

    const float4* h4 = (const float4*)h;
    for (int s = 0; s < TFRM; ++s) {
        int t = rev ? (TFRM - 1 - s) : s;
        const float* x = xs + (size_t)t * G4H;
        for (int r = tid; r < G4H; r += 1024) {
            const float4* w4 = (const float4*)(Whh + (size_t)r * HIDD);
            v2f a0 = {0.0f, 0.0f};
            v2f a1 = {0.0f, 0.0f};
#pragma unroll 4
            for (int j = 0; j < HIDD / 4; ++j) {
                float4 w = w4[j];
                float4 hh = h4[j];
                v2f wlo = {w.x, w.y}, whi = {w.z, w.w};
                v2f hlo = {hh.x, hh.y}, hhi = {hh.z, hh.w};
                a0 = pk_fma_f32(wlo, hlo, a0);
                a1 = pk_fma_f32(whi, hhi, a1);
            }
            g[r] = x[r] + a0.x + a0.y + a1.x + a1.y;
        }
        __syncthreads();
        for (int j = tid; j < HIDD; j += 1024) {
            float gi = 1.0f / (1.0f + expf(-g[j]));
            float gf = 1.0f / (1.0f + expf(-g[HIDD + j]));
            float gg = tanhf(g[2 * HIDD + j]);
            float go = 1.0f / (1.0f + expf(-g[3 * HIDD + j]));
            float cn = gf * c[j] + gi * gg;
            float hn = go * tanhf(cn);
            c[j] = cn; h[j] = hn;
            hs[(size_t)t * HIDD + j] = hn;
        }
        __syncthreads();
    }
}

// ------------------------------------------------------------------
static void launch_gemm(bool outbf16, bool relu,
                        const bf16* A, const bf16* B, const float* bias, void* C,
                        int M, int N, int K, int lda, int ldb, int ldc, hipStream_t s)
{
    dim3 grid(npad64(N) / 64, M / 32);
    if (outbf16) {
        if (relu) gemm_bias_wmma<true , true ><<<grid, 32, 0, s>>>(A, B, bias, C, N, K, lda, ldb, ldc);
        else      gemm_bias_wmma<true , false><<<grid, 32, 0, s>>>(A, B, bias, C, N, K, lda, ldb, ldc);
    } else {
        if (relu) gemm_bias_wmma<false, true ><<<grid, 32, 0, s>>>(A, B, bias, C, N, K, lda, ldb, ldc);
        else      gemm_bias_wmma<false, false><<<grid, 32, 0, s>>>(A, B, bias, C, N, K, lda, ldb, ldc);
    }
}

extern "C" void kernel_launch(void* const* d_in, const int* in_sizes, int n_in,
                              void* d_out, int out_size, void* d_ws, size_t ws_size,
                              hipStream_t stream)
{
    (void)in_sizes; (void)n_in; (void)out_size; (void)ws_size;
    // ---- input indices (setup_inputs dict order, params flattened) ----
    const float* ling    = (const float*)d_in[0];
    const float* ac      = (const float*)d_in[1];
    const float* mora    = (const float*)d_in[2];
    const float* eps     = (const float*)d_in[3];
    const float* fc11_W  = (const float*)d_in[4];
    const float* fc11_b  = (const float*)d_in[5];
    const float* l1_Wih_f = (const float*)d_in[6];
    const float* l1_Whh_f = (const float*)d_in[7];
    const float* l1_bih_f = (const float*)d_in[8];
    const float* l1_bhh_f = (const float*)d_in[9];
    const float* l1_Wih_b = (const float*)d_in[10];
    const float* l1_Whh_b = (const float*)d_in[11];
    const float* l1_bih_b = (const float*)d_in[12];
    const float* l1_bhh_b = (const float*)d_in[13];
    const float* fc21_W  = (const float*)d_in[14];
    const float* fc21_b  = (const float*)d_in[15];
    const float* fc22_W  = (const float*)d_in[16];
    const float* fc22_b  = (const float*)d_in[17];
    const float* fc12_W  = (const float*)d_in[18];
    const float* fc12_b  = (const float*)d_in[19];
    const float* a_Wih_f = (const float*)d_in[20];
    const float* a_Whh_f = (const float*)d_in[21];
    const float* a_bih_f = (const float*)d_in[22];
    const float* a_bhh_f = (const float*)d_in[23];
    const float* a_Wih_b = (const float*)d_in[24];
    const float* a_Whh_b = (const float*)d_in[25];
    const float* a_bih_b = (const float*)d_in[26];
    const float* a_bhh_b = (const float*)d_in[27];
    const float* b_Wih_f = (const float*)d_in[28];
    const float* b_Whh_f = (const float*)d_in[29];
    const float* b_bih_f = (const float*)d_in[30];
    const float* b_bhh_f = (const float*)d_in[31];
    const float* b_Wih_b = (const float*)d_in[32];
    const float* b_Whh_b = (const float*)d_in[33];
    const float* b_bih_b = (const float*)d_in[34];
    const float* b_bhh_b = (const float*)d_in[35];
    const float* fc3_W   = (const float*)d_in[36];
    const float* fc3_b   = (const float*)d_in[37];

    // ---- workspace carve-out ----
    char* wsp = (char*)d_ws;
    size_t off = 0;
    auto alloc = [&](size_t bytes) -> void* {
        void* p = wsp + off;
        off = (off + bytes + 255) & ~(size_t)255;
        return p;
    };
    bf16* A0    = (bf16*)alloc((size_t)TFRM * 832 * 2);     // padded A buffer (cat / x2)
    bf16* A1    = (bf16*)alloc((size_t)TFRM * 832 * 2);     // padded A buffer (fc11 / fc12 out)
    bf16* CC    = (bf16*)alloc((size_t)TFRM * H2D * 2);     // hs concat bf16
    bf16* Wc11  = (bf16*)alloc((size_t)npad64(CATD) * CATP * 2);
    bf16* W1f   = (bf16*)alloc((size_t)G4H * CATP * 2);
    bf16* W1b   = (bf16*)alloc((size_t)G4H * CATP * 2);
    bf16* Wc12  = (bf16*)alloc((size_t)npad64(X2D) * X2P * 2);
    bf16* W2af  = (bf16*)alloc((size_t)G4H * X2P * 2);
    bf16* W2ab  = (bf16*)alloc((size_t)G4H * X2P * 2);
    bf16* W2bf  = (bf16*)alloc((size_t)G4H * H2D * 2);
    bf16* W2bb  = (bf16*)alloc((size_t)G4H * H2D * 2);
    bf16* Wc3   = (bf16*)alloc((size_t)npad64(ACD) * H2D * 2);
    float* bs1f = (float*)alloc(G4H * 4);
    float* bs1b = (float*)alloc(G4H * 4);
    float* bsaf = (float*)alloc(G4H * 4);
    float* bsab = (float*)alloc(G4H * 4);
    float* bsbf = (float*)alloc(G4H * 4);
    float* bsbb = (float*)alloc(G4H * 4);
    float* xsf  = (float*)alloc((size_t)TFRM * G4H * 4);
    float* xsb  = (float*)alloc((size_t)TFRM * G4H * 4);
    float* h1f  = (float*)alloc((size_t)TFRM * HIDD * 4);
    float* h1b  = (float*)alloc((size_t)TFRM * HIDD * 4);
    float* h2f  = (float*)alloc((size_t)TFRM * HIDD * 4);
    float* h2b  = (float*)alloc((size_t)TFRM * HIDD * 4);
    int*   kcum = (int*)alloc(TFRM * 4);
    int*   idx  = (int*)alloc(NMOR * 4);
    float* zbuf = (float*)alloc((size_t)NMOR * ZD * 4);
    float* h3f = h1f;   // reuse after gather
    float* h3b = h1b;

    float* out_recon = (float*)d_out;                       // [8192,199]
    float* out_mu    = out_recon + (size_t)TFRM * ACD;      // [1024,16]
    float* out_lv    = out_mu + (size_t)NMOR * ZD;          // [1024,16]

    auto cvt = [&](const float* src, bf16* dst, int N, int K, int Kpad) {
        int total = npad64(N) * Kpad;
        k_cvt_pad<<<(total + 255) / 256, 256, 0, stream>>>(src, dst, N, K, npad64(N), Kpad);
    };
    auto addb = [&](const float* a, const float* b, float* o) {
        k_addbias<<<(G4H + 255) / 256, 256, 0, stream>>>(a, b, o, G4H);
    };

    // 1) weight conversions to padded bf16 (rows padded to x64 => no guards
    //    in the WMMA k-loop)
    cvt(fc11_W, Wc11, CATD, CATD, CATP);
    cvt(l1_Wih_f, W1f, G4H, CATD, CATP);
    cvt(l1_Wih_b, W1b, G4H, CATD, CATP);
    cvt(fc12_W, Wc12, X2D, X2D, X2P);
    cvt(a_Wih_f, W2af, G4H, X2D, X2P);
    cvt(a_Wih_b, W2ab, G4H, X2D, X2P);
    cvt(b_Wih_f, W2bf, G4H, H2D, H2D);
    cvt(b_Wih_b, W2bb, G4H, H2D, H2D);
    cvt(fc3_W, Wc3, ACD, H2D, H2D);
    addb(l1_bih_f, l1_bhh_f, bs1f);
    addb(l1_bih_b, l1_bhh_b, bs1b);
    addb(a_bih_f, a_bhh_f, bsaf);
    addb(a_bih_b, a_bhh_b, bsab);
    addb(b_bih_f, b_bhh_f, bsbf);
    addb(b_bih_b, b_bhh_b, bsbb);

    // 2) encoder: x = relu(fc11(cat(ling, ac)))
    k_concat_in<<<(TFRM * CATP + 255) / 256, 256, 0, stream>>>(ling, ac, A0);
    launch_gemm(true, true, A0, Wc11, fc11_b, A1, TFRM, CATD, CATP, CATP, CATP, CATP, stream);

    // 3) lstm1 input projections + scan
    launch_gemm(false, false, A1, W1f, bs1f, xsf, TFRM, G4H, CATP, CATP, CATP, G4H, stream);
    launch_gemm(false, false, A1, W1b, bs1b, xsb, TFRM, G4H, CATP, CATP, CATP, G4H, stream);
    k_lstm_scan<<<2, 1024, 0, stream>>>(xsf, xsb, l1_Whh_f, l1_Whh_b, h1f, h1b);

    // 4) mora gather -> mu/logvar -> z
    k_mora_scan<<<1, 1024, 0, stream>>>(mora, kcum, idx);
    k_gather_fc2<<<NMOR, 128, 0, stream>>>(h1f, h1b, idx, fc21_W, fc21_b, fc22_W, fc22_b,
                                           eps, zbuf, out_mu, out_lv);

    // 5) decoder: x2 = relu(fc12(cat(ling, z_tmp)))
    k_build_x2<<<(TFRM * X2P + 255) / 256, 256, 0, stream>>>(ling, zbuf, kcum, A0);
    launch_gemm(true, true, A0, Wc12, fc12_b, A1, TFRM, X2D, X2P, X2P, X2P, X2P, stream);

    // 6) lstm2a
    launch_gemm(false, false, A1, W2af, bsaf, xsf, TFRM, G4H, X2P, X2P, X2P, G4H, stream);
    launch_gemm(false, false, A1, W2ab, bsab, xsb, TFRM, G4H, X2P, X2P, X2P, G4H, stream);
    k_lstm_scan<<<2, 1024, 0, stream>>>(xsf, xsb, a_Whh_f, a_Whh_b, h2f, h2b);

    // 7) lstm2b (input = concat of lstm2a outputs, no relu)
    k_concat_hs<false><<<(TFRM * H2D + 255) / 256, 256, 0, stream>>>(h2f, h2b, CC);
    launch_gemm(false, false, CC, W2bf, bsbf, xsf, TFRM, G4H, H2D, H2D, H2D, G4H, stream);
    launch_gemm(false, false, CC, W2bb, bsbb, xsb, TFRM, G4H, H2D, H2D, H2D, G4H, stream);
    k_lstm_scan<<<2, 1024, 0, stream>>>(xsf, xsb, b_Whh_f, b_Whh_b, h3f, h3b);

    // 8) recon = fc3(relu(concat(h3)))
    k_concat_hs<true><<<(TFRM * H2D + 255) / 256, 256, 0, stream>>>(h3f, h3b, CC);
    launch_gemm(false, false, CC, Wc3, fc3_b, out_recon, TFRM, ACD, H2D, H2D, H2D, ACD, stream);
}
